// GNNEncoder_71184787964495
// MI455X (gfx1250) — matrix operand order repaired and checked by reading, not compiled, not run on hardware
//
#include <hip/hip_runtime.h>
#include <hip/hip_bf16.h>

typedef __attribute__((ext_vector_type(16))) _Float16 v16h;
typedef __attribute__((ext_vector_type(8)))  _Float16 v8h;
typedef __attribute__((ext_vector_type(8)))  float    v8f;
typedef int v4i __attribute__((vector_size(4 * sizeof(int))));

static constexpr int NN = 65536;
static constexpr int NE = 524288;
static constexpr int HID = 128;
static constexpr int MH  = 64;     // MLP_H
static constexpr int KS  = 16;     // K_SUPER
static constexpr int NG  = 8;      // N_GRAPHS

// ---------------------------------------------------------------------------
// helpers
// ---------------------------------------------------------------------------
__device__ __forceinline__ float siluf(float x) { return x / (1.f + __expf(-x)); }

// --- CDNA5 async memory->LDS copy (16B per lane), ASYNCcnt-tracked ----------
__device__ __forceinline__ void async_copy_b128(const _Float16* g, _Float16* l) {
#if __has_builtin(__builtin_amdgcn_global_load_async_to_lds_b128)
  typedef __attribute__((address_space(1))) v4i* gp_t;
  typedef __attribute__((address_space(3))) v4i* lp_t;
  __builtin_amdgcn_global_load_async_to_lds_b128(
      (gp_t)(uintptr_t)g, (lp_t)(unsigned)(uintptr_t)l, 0, 0);
#else
  unsigned lo = (unsigned)(uintptr_t)l;
  asm volatile("global_load_async_to_lds_b128 %0, %1, off"
               :: "v"(lo), "v"((const void*)g) : "memory");
#endif
}

__device__ __forceinline__ void wait_async_16() {
#if __has_builtin(__builtin_amdgcn_s_wait_asynccnt)
  __builtin_amdgcn_s_wait_asynccnt(16);
#else
  asm volatile("s_wait_asynccnt 0x10" ::: "memory");
#endif
}
__device__ __forceinline__ void wait_async_0() {
#if __has_builtin(__builtin_amdgcn_s_wait_asynccnt)
  __builtin_amdgcn_s_wait_asynccnt(0);
#else
  asm volatile("s_wait_asynccnt 0x0" ::: "memory");
#endif
}

// A operand (16xK f16, row-major in LDS, stride `stride` halfs), chunk kc (K=32).
// Layout (16-bit A 16x32, contiguous k): lane<16 -> M=lane, elems 0..15 = K0..15;
// lane>=16 -> K16..31 (matches load_B_pk).
__device__ __forceinline__ v16h load_A_lds(const _Float16* base, int stride, int kc, int lane) {
  int m    = lane & 15;
  int koff = (lane >> 4) << 4;          // 0 or 16
  const _Float16* p = base + m * stride + kc * 32 + koff;
  v8h lo = *(const v8h*)p;
  v8h hi = *(const v8h*)(p + 8);
  v16h a;
#pragma unroll
  for (int i = 0; i < 8; ++i) { a[i] = lo[i]; a[8 + i] = hi[i]; }
  return a;
}

// B operand tile packed as [n (16)][k (32)] halfs: lane<16 -> N=lane, K=0..15;
// lane>=16 -> N=lane-16, K=16..31 (contiguous per lane).
__device__ __forceinline__ v16h load_B_pk(const _Float16* tile, int lane) {
  return *(const v16h*)(tile + ((lane & 15) << 5) + ((lane >> 4) << 4));
}

// D = silu_or_id(A@B + bias) -> LDS (f16, stride OSTRIDE). NT tiles of 16 cols.
template<int NT, int OSTRIDE, bool ACT_SILU>
__device__ __forceinline__ void gemm_tile_store(const _Float16* A, int astride, int kchunks,
                                                const _Float16* Bpk, const float* bias,
                                                _Float16* outb, int lane) {
  int nl   = lane & 15;
  int moff = (lane >> 4) << 3;
  for (int nt = 0; nt < NT; ++nt) {
    float bv = bias[nt * 16 + nl];
    v8f c;
#pragma unroll
    for (int i = 0; i < 8; ++i) c[i] = bv;
    for (int kc = 0; kc < kchunks; ++kc) {
      v16h a = load_A_lds(A, astride, kc, lane);
      v16h b = load_B_pk(Bpk + (size_t)(kc * NT + nt) * 512, lane);
      c = __builtin_amdgcn_wmma_f32_16x16x32_f16(false, a, false, b, (short)0, c, false, false);
    }
#pragma unroll
    for (int r = 0; r < 8; ++r) {
      float v = c[r];
      if (ACT_SILU) v = siluf(v);
      outb[(moff + r) * OSTRIDE + nt * 16 + nl] = (_Float16)v;
    }
  }
}

// D = A@B + bias, atomically scattered to m_h[dst[row]*64 + col] (segment_sum).
template<int NT>
__device__ __forceinline__ void gemm_tile_atomic(const _Float16* A, int astride, int kchunks,
                                                 const _Float16* Bpk, const float* bias,
                                                 const int* dstb, float* m_h, int lane) {
  int nl   = lane & 15;
  int moff = (lane >> 4) << 3;
  for (int nt = 0; nt < NT; ++nt) {
    float bv = bias[nt * 16 + nl];
    v8f c;
#pragma unroll
    for (int i = 0; i < 8; ++i) c[i] = bv;
    for (int kc = 0; kc < kchunks; ++kc) {
      v16h a = load_A_lds(A, astride, kc, lane);
      v16h b = load_B_pk(Bpk + (size_t)(kc * NT + nt) * 512, lane);
      c = __builtin_amdgcn_wmma_f32_16x16x32_f16(false, a, false, b, (short)0, c, false, false);
    }
#pragma unroll
    for (int r = 0; r < 8; ++r) {
      int row = moff + r;
      atomicAdd(&m_h[(size_t)dstb[row] * MH + nt * 16 + nl], c[r]);
    }
  }
}

// D = relu(A@B + bias) -> LDS f32 (stride 128). Used by node phi_h L2.
template<int NT>
__device__ __forceinline__ void gemm_tile_relu_f32(const _Float16* A, int astride, int kchunks,
                                                   const _Float16* Bpk, const float* bias,
                                                   float* outb, int lane) {
  int nl   = lane & 15;
  int moff = (lane >> 4) << 3;
  for (int nt = 0; nt < NT; ++nt) {
    float bv = bias[nt * 16 + nl];
    v8f c;
#pragma unroll
    for (int i = 0; i < 8; ++i) c[i] = bv;
    for (int kc = 0; kc < kchunks; ++kc) {
      v16h a = load_A_lds(A, astride, kc, lane);
      v16h b = load_B_pk(Bpk + (size_t)(kc * NT + nt) * 512, lane);
      c = __builtin_amdgcn_wmma_f32_16x16x32_f16(false, a, false, b, (short)0, c, false, false);
    }
#pragma unroll
    for (int r = 0; r < 8; ++r)
      outb[(moff + r) * 128 + nt * 16 + nl] = fmaxf(c[r], 0.f);
  }
}

// ---------------------------------------------------------------------------
// weight packing: W (KxN f32, row-major) -> f16 tiles [kc][nc] of [n(16)][k(32)]
// ---------------------------------------------------------------------------
__global__ void pack_w(const float* __restrict__ W, _Float16* __restrict__ out,
                       int K, int N, int Nt) {
  int t  = blockIdx.x;
  int kc = t / Nt, nc = t % Nt;
  for (int idx = threadIdx.x; idx < 512; idx += blockDim.x) {
    int n = idx >> 5, k = idx & 31;
    int gk = kc * 32 + k, gn = nc * 16 + n;
    float v = (gk < K && gn < N) ? W[(size_t)gk * N + gn] : 0.f;
    out[(size_t)t * 512 + idx] = (_Float16)v;
  }
}

__global__ void zero_f32(float* p, int n) {
  int i = blockIdx.x * blockDim.x + threadIdx.x;
  if (i < n) p[i] = 0.f;
}

// ---------------------------------------------------------------------------
// layer-1 edge kernel (small din, f32 input): per wave 16 edges.
// ---------------------------------------------------------------------------
template<int CIN, int KPAD>
__global__ __launch_bounds__(128) void edge_kernel_l1(
    const float* __restrict__ xf,
    const float* __restrict__ x0, const int* __restrict__ ei, int nE,
    const _Float16* pe0, const float* be0,
    const _Float16* pe1, const float* be1,
    const _Float16* pe2, const float* be2,
    const _Float16* pv0, const float* bv0,
    const float* wv1, const float* bv1,
    float* __restrict__ m_h, float* __restrict__ m_v) {
  __shared__ _Float16 featS[4][16 * KPAD];
  __shared__ _Float16 h1S[4][16 * MH];
  __shared__ _Float16 h2S[4][16 * MH];
  __shared__ int dstS[4][16], srcS[4][16];
  __shared__ float relS[4][32];

  int wid = threadIdx.x >> 5, lane = threadIdx.x & 31;
  _Float16* feat = featS[wid];
  _Float16* h1 = h1S[wid];
  _Float16* h2 = h2S[wid];
  int* dstb = dstS[wid];
  int* srcb = srcS[wid];
  float* rel = relS[wid];

  int e0 = (blockIdx.x * 4 + wid) * 16;
  if (e0 >= nE) return;

  if (lane < 16) {
    int e = e0 + lane;
    int s = ei[e], d = ei[nE + e];
    srcb[lane] = s; dstb[lane] = d;
    float ps0 = x0[(size_t)s * 8 + 0], ps1 = x0[(size_t)s * 8 + 1];
    float vs0 = x0[(size_t)s * 8 + 2], vs1 = x0[(size_t)s * 8 + 3];
    float pd0 = x0[(size_t)d * 8 + 0], pd1 = x0[(size_t)d * 8 + 1];
    float vd0 = x0[(size_t)d * 8 + 2], vd1 = x0[(size_t)d * 8 + 3];
    float r0 = ps0 - pd0, r1 = ps1 - pd1;
    float w0 = vs0 - vd0, w1 = vs1 - vd1;
    rel[lane * 2 + 0] = r0; rel[lane * 2 + 1] = r1;
    feat[lane * KPAD + 2 * CIN]     = (_Float16)(r0 * r0 + r1 * r1);
    feat[lane * KPAD + 2 * CIN + 1] = (_Float16)(w0 * r0 + w1 * r1);
    for (int c = 2 * CIN + 2; c < KPAD; ++c) feat[lane * KPAD + c] = (_Float16)0.f;
  }

  for (int idx = lane; idx < 16 * 2 * CIN; idx += 32) {
    int ee = idx / (2 * CIN), c = idx % (2 * CIN);
    int node = (c < CIN) ? dstb[ee] : srcb[ee];
    int cc = (c < CIN) ? c : c - CIN;
    feat[ee * KPAD + c] = (_Float16)xf[(size_t)node * CIN + cc];
  }

  // phi_e: din -> 64 (silu) -> 64 (silu) -> 64, segment-sum to m_h[dst]
  gemm_tile_store<4, MH, true>(feat, KPAD, KPAD / 32, pe0, be0, h1, lane);
  gemm_tile_store<4, MH, true>(h1, MH, 2, pe1, be1, h2, lane);
  gemm_tile_atomic<4>(h2, MH, 2, pe2, be2, dstb, m_h, lane);

  // phi_v: din -> 64 (silu) -> 1, m_v_e = scalar * rel_pos
  gemm_tile_store<4, MH, true>(feat, KPAD, KPAD / 32, pv0, bv0, h1, lane);
  {
    int e = lane & 15, hf = lane >> 4;
    float part = 0.f;
    for (int j = 0; j < 32; ++j)
      part += (float)h1[e * MH + hf * 32 + j] * wv1[hf * 32 + j];
    part += __shfl_xor(part, 16, 32);
    if (lane < 16) {
      float sc = part + bv1[0];
      int d = dstb[e];
      atomicAdd(&m_v[(size_t)d * 2 + 0], sc * rel[e * 2 + 0]);
      atomicAdd(&m_v[(size_t)d * 2 + 1], sc * rel[e * 2 + 1]);
    }
  }
}

// ---------------------------------------------------------------------------
// layer-2 edge kernel: f16 node-state gather done with async mem->LDS copies,
// double-buffered and software-pipelined across 8 tiles per wave.
// 2 waves/block, 16 edges per tile.
// ---------------------------------------------------------------------------
__global__ __launch_bounds__(64) void edge_kernel_l2(
    const _Float16* __restrict__ xh,
    const float* __restrict__ x0, const int* __restrict__ ei, int nE,
    const _Float16* pe0, const float* be0,
    const _Float16* pe1, const float* be1,
    const _Float16* pe2, const float* be2,
    const _Float16* pv0, const float* bv0,
    const float* wv1, const float* bv1,
    float* __restrict__ m_h, float* __restrict__ m_v) {
  constexpr int CIN = 128, KPAD = 288, TPW = 8;
  __shared__ _Float16 featS[2][2][16 * KPAD];
  __shared__ _Float16 h1S[2][16 * MH];
  __shared__ _Float16 h2S[2][16 * MH];
  __shared__ int dstS[2][2][16], srcS[2][2][16];
  __shared__ float relS[2][2][32];

  int wid = threadIdx.x >> 5, lane = threadIdx.x & 31;
  int tile0 = (blockIdx.x * 2 + wid) * TPW;

  auto issue = [&](int t) {
    int buf = t & 1;
    int e0 = (tile0 + t) * 16;
    _Float16* feat = featS[wid][buf];
    int* db = dstS[wid][buf];
    int* sb = srcS[wid][buf];
    float* rl = relS[wid][buf];
    if (lane < 16) {
      int e = e0 + lane;
      int s = ei[e], d = ei[nE + e];
      sb[lane] = s; db[lane] = d;
      float ps0 = x0[(size_t)s * 8 + 0], ps1 = x0[(size_t)s * 8 + 1];
      float vs0 = x0[(size_t)s * 8 + 2], vs1 = x0[(size_t)s * 8 + 3];
      float pd0 = x0[(size_t)d * 8 + 0], pd1 = x0[(size_t)d * 8 + 1];
      float vd0 = x0[(size_t)d * 8 + 2], vd1 = x0[(size_t)d * 8 + 3];
      float r0 = ps0 - pd0, r1 = ps1 - pd1;
      float w0 = vs0 - vd0, w1 = vs1 - vd1;
      rl[lane * 2 + 0] = r0; rl[lane * 2 + 1] = r1;
      feat[lane * KPAD + 2 * CIN]     = (_Float16)(r0 * r0 + r1 * r1);
      feat[lane * KPAD + 2 * CIN + 1] = (_Float16)(w0 * r0 + w1 * r1);
      for (int c = 2 * CIN + 2; c < KPAD; ++c) feat[lane * KPAD + c] = (_Float16)0.f;
    }
    // async gather: 16 edges x 256 f16 (x_dst | x_src), 16B per op,
    // 16 ops per lane -> wave ASYNCcnt += 16
    for (int idx = lane; idx < 512; idx += 32) {
      int ee = idx >> 5, g = idx & 31, c8 = g * 8;
      int node = (c8 < CIN) ? db[ee] : sb[ee];
      int cc = c8 & (CIN - 1);
      async_copy_b128(xh + (size_t)node * CIN + cc, feat + ee * KPAD + c8);
    }
  };

  issue(0);
  for (int t = 0; t < TPW; ++t) {
    if (t + 1 < TPW) {
      issue(t + 1);       // prefetch next tile into other buffer
      wait_async_16();    // current tile's 16 async ops retired
    } else {
      wait_async_0();
    }
    int buf = t & 1;
    _Float16* feat = featS[wid][buf];
    int* db = dstS[wid][buf];
    float* rl = relS[wid][buf];
    _Float16* h1 = h1S[wid];
    _Float16* h2 = h2S[wid];

    gemm_tile_store<4, MH, true>(feat, KPAD, KPAD / 32, pe0, be0, h1, lane);
    gemm_tile_store<4, MH, true>(h1, MH, 2, pe1, be1, h2, lane);
    gemm_tile_atomic<4>(h2, MH, 2, pe2, be2, db, m_h, lane);

    gemm_tile_store<4, MH, true>(feat, KPAD, KPAD / 32, pv0, bv0, h1, lane);
    {
      int e = lane & 15, hf = lane >> 4;
      float part = 0.f;
      for (int j = 0; j < 32; ++j)
        part += (float)h1[e * MH + hf * 32 + j] * wv1[hf * 32 + j];
      part += __shfl_xor(part, 16, 32);
      if (lane < 16) {
        float sc = part + bv1[0];
        int d = db[e];
        atomicAdd(&m_v[(size_t)d * 2 + 0], sc * rl[e * 2 + 0]);
        atomicAdd(&m_v[(size_t)d * 2 + 1], sc * rl[e * 2 + 1]);
      }
    }
  }
}

// ---------------------------------------------------------------------------
// node kernel: g=[x, m_h, |m_v|] -> phi_h (64 silu, 128) -> relu -> layernorm
// writes f16 node state for the next stage.
// ---------------------------------------------------------------------------
template<int CIN, int KPAD, bool HALF_IN>
__global__ __launch_bounds__(128) void node_kernel(
    const float* __restrict__ xf, const _Float16* __restrict__ xh,
    const float* __restrict__ m_h, const float* __restrict__ m_v,
    const _Float16* pw0, const float* pb0,
    const _Float16* pw1, const float* pb1,
    const float* __restrict__ ln_g, const float* __restrict__ ln_b,
    _Float16* __restrict__ h_out) {
  __shared__ __align__(16) char uS[4][16 * 128 * 4];   // union: feat f16 / out f32
  __shared__ _Float16 hbS[4][16 * MH];
  __shared__ float miS[4][32];

  int wid = threadIdx.x >> 5, lane = threadIdx.x & 31;
  _Float16* feat = (_Float16*)uS[wid];
  float* outf = (float*)uS[wid];
  _Float16* hb = hbS[wid];
  float* mi = miS[wid];

  int n0 = (blockIdx.x * 4 + wid) * 16;
  if (n0 >= NN) return;

  if (HALF_IN) {
    const int GPE = CIN / 8;
    for (int idx = lane; idx < 16 * GPE; idx += 32) {
      int ee = idx / GPE, g = idx % GPE;
      v8h v = *(const v8h*)(xh + (size_t)(n0 + ee) * CIN + g * 8);
      *(v8h*)(feat + ee * KPAD + g * 8) = v;
    }
  } else {
    for (int idx = lane; idx < 16 * CIN; idx += 32) {
      int ee = idx / CIN, c = idx % CIN;
      feat[ee * KPAD + c] = (_Float16)xf[(size_t)(n0 + ee) * CIN + c];
    }
  }
  for (int idx = lane; idx < 16 * MH; idx += 32) {
    int ee = idx >> 6, c = idx & 63;
    feat[ee * KPAD + CIN + c] = (_Float16)m_h[(size_t)(n0 + ee) * MH + c];
  }
  if (lane < 16) {
    int n = n0 + lane;
    float a = m_v[(size_t)n * 2 + 0], b = m_v[(size_t)n * 2 + 1];
    feat[lane * KPAD + CIN + MH] = (_Float16)sqrtf(a * a + b * b + 1e-12f);
    for (int c = CIN + MH + 1; c < KPAD; ++c) feat[lane * KPAD + c] = (_Float16)0.f;
  }

  gemm_tile_store<4, MH, true>(feat, KPAD, KPAD / 32, pw0, pb0, hb, lane);
  gemm_tile_relu_f32<8>(hb, MH, 2, pw1, pb1, outf, lane);   // overwrites feat (dead)

  if (lane < 16) {
    float s = 0.f, s2 = 0.f;
    for (int c = 0; c < 128; ++c) { float v = outf[lane * 128 + c]; s += v; s2 += v * v; }
    float m = s * (1.f / 128.f);
    float var = s2 * (1.f / 128.f) - m * m;
    mi[lane * 2 + 0] = m;
    mi[lane * 2 + 1] = rsqrtf(var + 1e-5f);
  }
  for (int idx = lane; idx < 16 * 128; idx += 32) {
    int ee = idx >> 7, c = idx & 127;
    float v = (outf[idx] - mi[ee * 2]) * mi[ee * 2 + 1] * ln_g[c] + ln_b[c];
    h_out[(size_t)(n0 + ee) * 128 + c] = (_Float16)v;
  }
}

// ---------------------------------------------------------------------------
// pooling: logits via WMMA (128->16), softmax, entropy, per-graph s^T@h via
// masked WMMA tiles + ds_add_f32 block reduction + global atomics.
// ---------------------------------------------------------------------------
__global__ __launch_bounds__(128) void pool_kernel(
    const _Float16* __restrict__ h, const float* __restrict__ x0,
    const int* __restrict__ batch,
    const _Float16* pw_pool, const float* pb_pool,
    float* __restrict__ num, float* __restrict__ den,
    float* __restrict__ munum, float* __restrict__ ent_g,
    float* __restrict__ s_out) {
  __shared__ _Float16 hbS[4][32 * 128];
  __shared__ _Float16 slS[4][32 * 16];
  __shared__ _Float16 alS[4][32 * 16];
  __shared__ float ltS[4][16 * 16];
  __shared__ float acc[16 * 128];
  __shared__ float dacc[16];
  __shared__ float macc[32];
  __shared__ int bmm[2];

  int wid = threadIdx.x >> 5, lane = threadIdx.x & 31;
  _Float16* hb = hbS[wid];
  _Float16* sl = slS[wid];
  _Float16* al = alS[wid];
  float* lt = ltS[wid];

  int n0 = blockIdx.x * 128 + wid * 32;

  for (int idx = lane; idx < 32 * 16; idx += 32) {
    int ee = idx >> 4, g = idx & 15;
    *(v8h*)&hb[ee * 128 + g * 8] = *(const v8h*)&h[(size_t)(n0 + ee) * 128 + g * 8];
  }
  int node = n0 + lane;
  int bt = batch[node];
  float p0 = x0[(size_t)node * 8 + 0], p1 = x0[(size_t)node * 8 + 1];

  int nl = lane & 15, moff = (lane >> 4) << 3;
  float ent_local = 0.f;
  for (int sub = 0; sub < 2; ++sub) {
    const _Float16* Af = hb + sub * 16 * 128;
    float bv = pb_pool[nl];
    v8f c;
#pragma unroll
    for (int i = 0; i < 8; ++i) c[i] = bv;
    for (int kc = 0; kc < 4; ++kc) {
      v16h a = load_A_lds(Af, 128, kc, lane);
      v16h b = load_B_pk(pw_pool + (size_t)kc * 512, lane);
      c = __builtin_amdgcn_wmma_f32_16x16x32_f16(false, a, false, b, (short)0, c, false, false);
    }
#pragma unroll
    for (int r = 0; r < 8; ++r) lt[(moff + r) * 16 + nl] = c[r];

    if (lane < 16) {
      float mx = -3.4e38f;
      for (int k = 0; k < 16; ++k) mx = fmaxf(mx, lt[lane * 16 + k]);
      float sum = 0.f;
      for (int k = 0; k < 16; ++k) sum += __expf(lt[lane * 16 + k] - mx);
      float inv = 1.f / sum;
      int nn = n0 + sub * 16 + lane;
      for (int k = 0; k < 16; ++k) {
        float sv = __expf(lt[lane * 16 + k] - mx) * inv;
        sl[(sub * 16 + lane) * 16 + k] = (_Float16)sv;
        s_out[(size_t)nn * 16 + k] = sv;
        ent_local += sv * __logf(sv + 1e-9f);
      }
    }
  }

  if (threadIdx.x == 0) bmm[0] = batch[blockIdx.x * 128];
  if (threadIdx.x == 1) bmm[1] = batch[blockIdx.x * 128 + 127];
  __syncthreads();
  int bmin = bmm[0], bmax = bmm[1];

  for (int b = bmin; b <= bmax; ++b) {
    for (int idx = threadIdx.x; idx < 2048; idx += 128) acc[idx] = 0.f;
    if (threadIdx.x < 16) dacc[threadIdx.x] = 0.f;
    if (threadIdx.x < 32) macc[threadIdx.x] = 0.f;
    __syncthreads();

    bool mine = (bt == b);
#pragma unroll
    for (int k = 0; k < 16; ++k)
      al[lane * 16 + k] = mine ? sl[lane * 16 + k] : (_Float16)0.f;
    if (mine) {
      for (int k = 0; k < 16; ++k) {
        float sv = (float)sl[lane * 16 + k];
        atomicAdd(&dacc[k], sv);
        atomicAdd(&macc[k * 2 + 0], sv * p0);
        atomicAdd(&macc[k * 2 + 1], sv * p1);
      }
    }

    // A[M=k][K=node(32)] from al; B[K=node][N=col] from hb
    v16h a;
#pragma unroll
    for (int j = 0; j < 16; ++j) {
      int K = j + ((lane >> 4) << 4);
      a[j] = al[K * 16 + nl];
    }
    for (int nt = 0; nt < 8; ++nt) {
      v16h bvv;
#pragma unroll
      for (int j = 0; j < 16; ++j) {
        int nj = j + ((lane >> 4) << 4);
        bvv[j] = hb[nj * 128 + nt * 16 + nl];
      }
      v8f c;
#pragma unroll
      for (int i = 0; i < 8; ++i) c[i] = 0.f;
      c = __builtin_amdgcn_wmma_f32_16x16x32_f16(false, a, false, bvv, (short)0, c, false, false);
#pragma unroll
      for (int r = 0; r < 8; ++r)
        atomicAdd(&acc[(moff + r) * 128 + nt * 16 + nl], c[r]);
    }
    __syncthreads();
    for (int idx = threadIdx.x; idx < 2048; idx += 128)
      atomicAdd(&num[(size_t)b * 2048 + idx], acc[idx]);
    if (threadIdx.x < 16) atomicAdd(&den[b * 16 + threadIdx.x], dacc[threadIdx.x]);
    if (threadIdx.x < 32) atomicAdd(&munum[b * 32 + threadIdx.x], macc[threadIdx.x]);
    __syncthreads();
  }

  float e = ent_local;
#pragma unroll
  for (int off = 16; off >= 1; off >>= 1) e += __shfl_xor(e, off, 32);
  if (lane == 0) atomicAdd(ent_g, e);
}

// ---------------------------------------------------------------------------
// finalize: pooled=num/den, mu, out-MLP (128->128 relu ->64), loss
// ---------------------------------------------------------------------------
__global__ __launch_bounds__(128) void finalize_kernel(
    const float* __restrict__ num, const float* __restrict__ den,
    const float* __restrict__ munum, const float* __restrict__ ent_g,
    const float* __restrict__ O0w, const float* __restrict__ O0b,
    const float* __restrict__ O1w, const float* __restrict__ O1b,
    float* __restrict__ out_latent, float* __restrict__ out_mu,
    float* __restrict__ out_loss) {
  int b = blockIdx.x, t = threadIdx.x;
  __shared__ float pooled[16 * 128];
  __shared__ float hid[16 * 128];
  for (int idx = t; idx < 2048; idx += 128) {
    int k = idx >> 7;
    pooled[idx] = num[(size_t)b * 2048 + idx] / (den[b * 16 + k] + 1e-8f);
  }
  if (t < 32) {
    int k = t >> 1;
    out_mu[b * 32 + t] = munum[b * 32 + t] / (den[b * 16 + k] + 1e-8f);
  }
  __syncthreads();
  for (int k = 0; k < 16; ++k) {
    float s = O0b[t];
    for (int c = 0; c < 128; ++c) s += pooled[k * 128 + c] * O0w[c * 128 + t];
    hid[k * 128 + t] = fmaxf(s, 0.f);
  }
  __syncthreads();
  for (int idx = t; idx < 1024; idx += 128) {
    int k = idx >> 6, o = idx & 63;
    float s = O1b[o];
    for (int hh = 0; hh < 128; ++hh) s += hid[k * 128 + hh] * O1w[hh * 64 + o];
    out_latent[(size_t)b * 1024 + idx] = s;
  }
  if (b == 0 && t == 0) out_loss[0] = -ent_g[0] / (float)NN;
}

// ---------------------------------------------------------------------------
// launch
// ---------------------------------------------------------------------------
enum {
  IN_X = 0, IN_EI, IN_BATCH,
  G1_E_W0, G1_E_B0, G1_E_W1, G1_E_B1, G1_E_W2, G1_E_B2,
  G1_H_W0, G1_H_B0, G1_H_W1, G1_H_B1,
  G1_V_W0, G1_V_B0, G1_V_W1, G1_V_B1,
  G2_E_W0, G2_E_B0, G2_E_W1, G2_E_B1, G2_E_W2, G2_E_B2,
  G2_H_W0, G2_H_B0, G2_H_W1, G2_H_B1,
  G2_V_W0, G2_V_B0, G2_V_W1, G2_V_B1,
  LN1_G, LN1_B, LN2_G, LN2_B,
  OUT_W0, OUT_B0, OUT_W1, OUT_B1,
  POOL_W, POOL_B
};

// packed weight sub-offsets in halfs
static constexpr size_t PK_L1E0 = 0;
static constexpr size_t PK_L1E1 = PK_L1E0 + 4 * 512;
static constexpr size_t PK_L1E2 = PK_L1E1 + 8 * 512;
static constexpr size_t PK_L1V0 = PK_L1E2 + 8 * 512;
static constexpr size_t PK_L1H0 = PK_L1V0 + 4 * 512;
static constexpr size_t PK_L1H1 = PK_L1H0 + 12 * 512;
static constexpr size_t PK_L2E0 = PK_L1H1 + 16 * 512;
static constexpr size_t PK_L2E1 = PK_L2E0 + 36 * 512;
static constexpr size_t PK_L2E2 = PK_L2E1 + 8 * 512;
static constexpr size_t PK_L2V0 = PK_L2E2 + 8 * 512;
static constexpr size_t PK_L2H0 = PK_L2V0 + 36 * 512;
static constexpr size_t PK_L2H1 = PK_L2H0 + 28 * 512;
static constexpr size_t PK_POOL = PK_L2H1 + 16 * 512;
static constexpr size_t PK_TOT  = PK_POOL + 4 * 512;

extern "C" void kernel_launch(void* const* d_in, const int* in_sizes, int n_in,
                              void* d_out, int out_size, void* d_ws, size_t ws_size,
                              hipStream_t stream) {
  (void)in_sizes; (void)n_in; (void)out_size; (void)ws_size;
  char* ws = (char*)d_ws;
  size_t off = 0;
  auto take = [&](size_t bytes) -> size_t {
    size_t o = off; off += (bytes + 255) & ~(size_t)255; return o;
  };
  size_t o_h1  = take((size_t)NN * 128 * 2);
  size_t o_h2  = take((size_t)NN * 128 * 2);
  size_t o_mh  = take((size_t)NN * MH * 4);
  size_t o_mv  = take((size_t)NN * 2 * 4);
  size_t o_num = take((size_t)NG * 2048 * 4);
  size_t o_den = take((size_t)NG * 16 * 4);
  size_t o_mun = take((size_t)NG * 32 * 4);
  size_t o_ent = take(256);
  size_t o_pk  = take(PK_TOT * 2);

  _Float16* h1 = (_Float16*)(ws + o_h1);
  _Float16* h2 = (_Float16*)(ws + o_h2);
  float* mh  = (float*)(ws + o_mh);
  float* mv  = (float*)(ws + o_mv);
  float* num = (float*)(ws + o_num);
  float* den = (float*)(ws + o_den);
  float* mun = (float*)(ws + o_mun);
  float* ent = (float*)(ws + o_ent);
  _Float16* pk = (_Float16*)(ws + o_pk);

  auto F = [&](int i) { return (const float*)d_in[i]; };
  const float* x0    = F(IN_X);
  const int*   ei    = (const int*)d_in[IN_EI];
  const int*   batch = (const int*)d_in[IN_BATCH];
  float* out = (float*)d_out;
  float* out_latent = out;
  float* out_s      = out + 8192;
  float* out_loss   = out + 8192 + (size_t)NN * 16;
  float* out_mu     = out_loss + 1;

  // --- pack weights (f32 -> f16 WMMA-B tiles) ---
  pack_w<<<4,  256, 0, stream>>>(F(G1_E_W0), pk + PK_L1E0,  18,  64, 4);
  pack_w<<<8,  256, 0, stream>>>(F(G1_E_W1), pk + PK_L1E1,  64,  64, 4);
  pack_w<<<8,  256, 0, stream>>>(F(G1_E_W2), pk + PK_L1E2,  64,  64, 4);
  pack_w<<<4,  256, 0, stream>>>(F(G1_V_W0), pk + PK_L1V0,  18,  64, 4);
  pack_w<<<12, 256, 0, stream>>>(F(G1_H_W0), pk + PK_L1H0,  73,  64, 4);
  pack_w<<<16, 256, 0, stream>>>(F(G1_H_W1), pk + PK_L1H1,  64, 128, 8);
  pack_w<<<36, 256, 0, stream>>>(F(G2_E_W0), pk + PK_L2E0, 258,  64, 4);
  pack_w<<<8,  256, 0, stream>>>(F(G2_E_W1), pk + PK_L2E1,  64,  64, 4);
  pack_w<<<8,  256, 0, stream>>>(F(G2_E_W2), pk + PK_L2E2,  64,  64, 4);
  pack_w<<<36, 256, 0, stream>>>(F(G2_V_W0), pk + PK_L2V0, 258,  64, 4);
  pack_w<<<28, 256, 0, stream>>>(F(G2_H_W0), pk + PK_L2H0, 193,  64, 4);
  pack_w<<<16, 256, 0, stream>>>(F(G2_H_W1), pk + PK_L2H1,  64, 128, 8);
  pack_w<<<4,  256, 0, stream>>>(F(POOL_W),  pk + PK_POOL, 128,  16, 1);

  auto zero = [&](float* p, int n) {
    zero_f32<<<(n + 255) / 256, 256, 0, stream>>>(p, n);
  };
  zero(mh, NN * MH);
  zero(mv, NN * 2);
  zero(num, NG * 2048);
  zero(den, NG * 16);
  zero(mun, NG * 32);
  zero(ent, 1);

  // --- layer 1 ---
  edge_kernel_l1<8, 32><<<NE / 64, 128, 0, stream>>>(
      x0, x0, ei, NE,
      pk + PK_L1E0, F(G1_E_B0), pk + PK_L1E1, F(G1_E_B1), pk + PK_L1E2, F(G1_E_B2),
      pk + PK_L1V0, F(G1_V_B0), F(G1_V_W1), F(G1_V_B1), mh, mv);
  node_kernel<8, 96, false><<<NN / 64, 128, 0, stream>>>(
      x0, nullptr, mh, mv,
      pk + PK_L1H0, F(G1_H_B0), pk + PK_L1H1, F(G1_H_B1),
      F(LN1_G), F(LN1_B), h1);

  zero(mh, NN * MH);
  zero(mv, NN * 2);

  // --- layer 2 (async-pipelined gather) ---
  edge_kernel_l2<<<NE / 256, 64, 0, stream>>>(
      h1, x0, ei, NE,
      pk + PK_L2E0, F(G2_E_B0), pk + PK_L2E1, F(G2_E_B1), pk + PK_L2E2, F(G2_E_B2),
      pk + PK_L2V0, F(G2_V_B0), F(G2_V_W1), F(G2_V_B1), mh, mv);
  node_kernel<128, 224, true><<<NN / 64, 128, 0, stream>>>(
      nullptr, h1, mh, mv,
      pk + PK_L2H0, F(G2_H_B0), pk + PK_L2H1, F(G2_H_B1),
      F(LN2_G), F(LN2_B), h2);

  // --- pooling + outputs ---
  pool_kernel<<<NN / 128, 128, 0, stream>>>(
      h2, x0, batch, pk + PK_POOL, F(POOL_B), num, den, mun, ent, out_s);
  finalize_kernel<<<NG, 128, 0, stream>>>(
      num, den, mun, ent, F(OUT_W0), F(OUT_B0), F(OUT_W1), F(OUT_B1),
      out_latent, out_mu, out_loss);
}
